// HigherOrderTimeDistributedDense_71356586656491
// MI455X (gfx1250) — matrix operand order repaired and checked by reading, not compiled, not run on hardware
//
#include <hip/hip_runtime.h>
#include <cmath>

// ---------------------------------------------------------------------------
// out[b,t1,t2,o] = tanh( sum_d X[b,t1,t2,d] * W[d,o] + b[o] )
// Flattened: C(65536x512) = tanh( A(65536x512) * W(512x512) + b )
// bf16 hi/lo split GEMM on v_wmma_f32_16x16x32_bf16 (3 products -> ~f32 acc).
// ---------------------------------------------------------------------------

typedef __attribute__((ext_vector_type(8)))  float  v8f;
typedef __attribute__((ext_vector_type(8)))  __bf16 v8bf;
typedef __attribute__((ext_vector_type(16))) __bf16 v16bf;

constexpr int Mtot = 8 * 64 * 128;   // 65536
constexpr int Kdim = 512;
constexpr int Ndim = 512;

constexpr int BM = 128;
constexpr int BN = 128;
constexpr int BK = 32;
constexpr int KSTEPS = Kdim / BK;    // 16
constexpr int LDA = BK + 8;          // 40 bf16 -> 80B row (16B aligned, padded)
constexpr int LDB = BK + 8;          // Wt[n][k] stride

__device__ __forceinline__ unsigned short bf16_rn(float f) {
  unsigned u = __float_as_uint(f);
  u += 0x7FFFu + ((u >> 16) & 1u);   // round-to-nearest-even
  return (unsigned short)(u >> 16);
}
__device__ __forceinline__ float bf16_f32(unsigned short h) {
  return __uint_as_float(((unsigned)h) << 16);
}

__device__ __forceinline__ v16bf ld_frag(const unsigned short* p0,
                                         const unsigned short* p1) {
  v8bf a = *(const v8bf*)p0;   // 16B-aligned -> ds_load_b128
  v8bf b = *(const v8bf*)p1;
  return __builtin_shufflevector(a, b, 0, 1, 2, 3, 4, 5, 6, 7,
                                       8, 9, 10, 11, 12, 13, 14, 15);
}

__global__ __launch_bounds__(256)
void dense_tanh_wmma(const float* __restrict__ X,
                     const float* __restrict__ W,
                     const float* __restrict__ bvec,
                     float* __restrict__ out) {
  __shared__ unsigned short sAh[BM * LDA];
  __shared__ unsigned short sAl[BM * LDA];
  __shared__ unsigned short sBh[BN * LDB];   // transposed: Wt[n][k]
  __shared__ unsigned short sBl[BN * LDB];

  const int tid = threadIdx.x;
  const int wgN = blockIdx.x & 3;            // Ndim/BN = 4
  const int wgM = blockIdx.x >> 2;           // Mtot/BM = 512
  const int m0  = wgM * BM;
  const int n0  = wgN * BN;

  const int wave  = tid >> 5;                // 8 waves
  const int lane  = tid & 31;
  const int l15   = lane & 15;
  const int half  = lane >> 4;               // 0/1 (K-half selector)
  const int waveM = wave & 1;                // 2 waves along M  (64 rows each)
  const int waveN = wave >> 1;               // 4 waves along N  (32 cols each)

  v8f acc[4][2];
#pragma unroll
  for (int mt = 0; mt < 4; ++mt)
#pragma unroll
    for (int nt = 0; nt < 2; ++nt)
      acc[mt][nt] = (v8f){0.f, 0.f, 0.f, 0.f, 0.f, 0.f, 0.f, 0.f};

  // staging coordinates
  const int aCol4 = (tid & 7) * 4;           // 0..28 step 4 (K offset, float4)
  const int aRow  = tid >> 3;                // 0..31
  const int bLane = tid & 31;                // N sub-index
  const int bK    = tid >> 5;                // 0..7 (K row)

  for (int kt = 0; kt < KSTEPS; ++kt) {
    const int k0 = kt * BK;

    // -------- stage A: 128x32 f32 -> bf16 hi/lo, row-major [m][k] --------
#pragma unroll
    for (int it = 0; it < 4; ++it) {
      const int row = aRow + it * 32;
      const float4 v = *(const float4*)(X + (size_t)(m0 + row) * Kdim + k0 + aCol4);
      unsigned short h0 = bf16_rn(v.x), h1 = bf16_rn(v.y),
                     h2 = bf16_rn(v.z), h3 = bf16_rn(v.w);
      unsigned short l0 = bf16_rn(v.x - bf16_f32(h0)),
                     l1 = bf16_rn(v.y - bf16_f32(h1)),
                     l2 = bf16_rn(v.z - bf16_f32(h2)),
                     l3 = bf16_rn(v.w - bf16_f32(h3));
      ushort4 ph; ph.x = h0; ph.y = h1; ph.z = h2; ph.w = h3;
      ushort4 pl; pl.x = l0; pl.y = l1; pl.z = l2; pl.w = l3;
      *(ushort4*)&sAh[row * LDA + aCol4] = ph;   // ds_store_b64
      *(ushort4*)&sAl[row * LDA + aCol4] = pl;
    }

    // ---- stage W: 32x128 f32, transpose -> Wt[n][k] bf16 hi/lo ----
#pragma unroll
    for (int it = 0; it < 4; ++it) {
      const int k = bK + it * 8;
      const float* wrow = W + (size_t)(k0 + k) * Ndim + n0;
#pragma unroll
      for (int j = 0; j < 4; ++j) {
        const int n = bLane + 32 * j;
        const float v = wrow[n];
        const unsigned short hh = bf16_rn(v);
        const unsigned short ll = bf16_rn(v - bf16_f32(hh));
        sBh[n * LDB + k] = hh;
        sBl[n * LDB + k] = ll;
      }
    }

    // prefetch next K-tile while this one is consumed
    if (kt + 1 < KSTEPS) {
      __builtin_prefetch((const void*)(X + (size_t)(m0 + aRow) * Kdim + k0 + BK + aCol4), 0, 0);
      __builtin_prefetch((const void*)(W + (size_t)(k0 + BK + bK) * Ndim + n0 + bLane * 4), 0, 0);
    }

    __syncthreads();

    // -------- compute: 4x2 tiles of 16x16, K=32 per step --------
    // A fragment layout (16-bit A 16x32): lane m=l15; VGPR0-3 K=half*8+0..7,
    // VGPR4-7 K=16+half*8+0..7  -> two b128 loads at +0 and +16 elements.
    v16bf ah[4], al[4];
#pragma unroll
    for (int mt = 0; mt < 4; ++mt) {
      const int mrow = waveM * 64 + mt * 16 + l15;
      const unsigned short* pH = &sAh[mrow * LDA + half * 8];
      const unsigned short* pL = &sAl[mrow * LDA + half * 8];
      ah[mt] = ld_frag(pH, pH + 16);
      al[mt] = ld_frag(pL, pL + 16);
    }

#pragma unroll
    for (int nt = 0; nt < 2; ++nt) {
      // B fragment (32x16): lane n=l15; VGPR0-3 K=half*8+0..7,
      // VGPR4-7 K=16+half*8+0..7 in Wt[n][k] -> two b128 loads.
      const int nrow = waveN * 32 + nt * 16 + l15;
      const unsigned short* qH = &sBh[nrow * LDB + half * 8];
      const unsigned short* qL = &sBl[nrow * LDB + half * 8];
      const v16bf bh = ld_frag(qH, qH + 16);
      const v16bf bl = ld_frag(qL, qL + 16);
#pragma unroll
      for (int mt = 0; mt < 4; ++mt) {
        acc[mt][nt] = __builtin_amdgcn_wmma_f32_16x16x32_bf16(
            false, ah[mt], false, bh, (short)0, acc[mt][nt], false, false);
        acc[mt][nt] = __builtin_amdgcn_wmma_f32_16x16x32_bf16(
            false, al[mt], false, bh, (short)0, acc[mt][nt], false, false);
        acc[mt][nt] = __builtin_amdgcn_wmma_f32_16x16x32_bf16(
            false, ah[mt], false, bl, (short)0, acc[mt][nt], false, false);
      }
    }
    __syncthreads();
  }

  // -------- epilogue: bias + tanh, f32 store --------
  // C/D layout: VGPR r holds element (M = r + 8*half, N = l15).
#pragma unroll
  for (int nt = 0; nt < 2; ++nt) {
    const int col = n0 + waveN * 32 + nt * 16 + l15;
    const float bv = bvec[col];
#pragma unroll
    for (int mt = 0; mt < 4; ++mt) {
      const int row0 = m0 + waveM * 64 + mt * 16 + half * 8;
#pragma unroll
      for (int r = 0; r < 8; ++r) {
        const float v = acc[mt][nt][r] + bv;
        out[(size_t)(row0 + r) * Ndim + col] = tanhf(v);
      }
    }
  }
}

extern "C" void kernel_launch(void* const* d_in, const int* in_sizes, int n_in,
                              void* d_out, int out_size, void* d_ws, size_t ws_size,
                              hipStream_t stream) {
  const float* X = (const float*)d_in[0];   // (8,64,128,512) f32
  const float* W = (const float*)d_in[1];   // (512,512) f32
  const float* b = (const float*)d_in[2];   // (512,) f32
  float* out = (float*)d_out;               // (8,64,128,512) f32

  dim3 grid((Mtot / BM) * (Ndim / BN));     // 512 * 4 = 2048 workgroups
  dim3 block(256);                          // 8 wave32s
  hipLaunchKernelGGL(dense_tanh_wmma, grid, block, 0, stream, X, W, b, out);
}
